// CustomMSDeformableAttentionLite_62517543960682
// MI455X (gfx1250) — matrix operand order repaired and checked by reading, not compiled, not run on hardware
//
#include <hip/hip_runtime.h>
#include <hip/hip_bf16.h>
#include <math.h>

typedef __attribute__((ext_vector_type(16))) _Float16 v16h;
typedef __attribute__((ext_vector_type(8)))  float    v8f;
typedef __attribute__((ext_vector_type(4)))  unsigned int v4u;
typedef __attribute__((ext_vector_type(8)))  int      v8i;
typedef __attribute__((ext_vector_type(4)))  int      v4i;

#define B_  8
#define Q_  900
#define E_  256
#define NH_ 8
#define NL_ 4
#define NP_ 4
#define HD_ 32
#define S_  19560

// ---------------------------------------------------------------------------
// TDM: DMA a [rows x 32] f32 tile (row stride K elements) from global memory
// into LDS at lds_byte_addr, padding each 32-DWORD row with 4 DWORDs so the
// LDS row stride is 36 floats (144B: 16B-aligned, 36*m mod 64 distinct for
// m=0..15 -> conflict-free fragment reads). Rows >= rows_valid are zero-filled
// by TDM OOB handling. Tracked by TENSORcnt.
// ---------------------------------------------------------------------------
__device__ __forceinline__
void tdm_load_a_tile(const float* gptr, unsigned lds_byte_addr,
                     int rows_valid, int K)
{
    unsigned long long ga = (unsigned long long)gptr;
    v4u g0;
    g0[0] = 1u;                                                  // count=1, user mode
    g0[1] = lds_byte_addr;                                       // lds_addr
    g0[2] = (unsigned)(ga & 0xffffffffu);                        // global_addr lo
    g0[3] = (unsigned)((ga >> 32) & 0x01ffffffu) | (2u << 30);   // global_addr hi | type=2

    const unsigned w0 = (2u << 16)   // data_size = 4 bytes
                      | (1u << 20)   // pad_enable
                      | (4u << 22)   // pad_interval code 4 = 32 DWORDs
                      | (3u << 25);  // pad_amount  code 3 = 4 DWORDs
    const unsigned td0 = 32u;                  // tensor_dim0 (tile fully in-bounds in x)
    const unsigned td1 = (unsigned)rows_valid; // tensor_dim1 (tail rows -> zero fill)
    const unsigned tile0 = 32u, tile1 = 128u;
    const unsigned long long s0 = (unsigned long long)(unsigned)K; // dim0 stride

    v8i g1;
    g1[0] = (int)w0;
    g1[1] = (int)((td0 & 0xffffu) << 16);                              // barrier_addr=0 | td0[15:0]
    g1[2] = (int)(((td0 >> 16) & 0xffffu) | ((td1 & 0xffffu) << 16));  // td0[31:16] | td1[15:0]
    g1[3] = (int)(((td1 >> 16) & 0xffffu) | (tile0 << 16));            // td1[31:16] | tile_dim0
    g1[4] = (int)tile1;                                                // tile_dim1 | tile_dim2=0
    g1[5] = (int)(unsigned)(s0 & 0xffffffffu);                         // stride0 lo
    g1[6] = (int)(unsigned)((s0 >> 32) & 0xffffu);                     // stride0 hi | stride1 lo=0
    g1[7] = 0;

    v4i gz = {0, 0, 0, 0};
#if defined(__clang_major__) && (__clang_major__ >= 23)
    v8i gz8 = {0, 0, 0, 0, 0, 0, 0, 0};
    __builtin_amdgcn_tensor_load_to_lds(g0, g1, gz, gz, gz8, 0);
#else
    __builtin_amdgcn_tensor_load_to_lds(g0, g1, gz, gz, 0);
#endif
}

// ---------------------------------------------------------------------------
// WMMA GEMM: C[M,N] = A[M,K] * Bm[K,N] + bias[N] (+ residual[M,N])
// fp32 inputs, f16 fragments, fp32 WMMA accumulation, OutT output (f32/f16).
// Block: 256 threads (8 waves). Block tile 128(M) x 16(N). K step 32.
// A tiles move via the Tensor Data Mover (double-buffered, async, hardware
// row padding); B tiles are staged manually (need f32->f16 conversion).
// One barrier per K-step.
// ---------------------------------------------------------------------------
template <typename OutT>
__global__ __launch_bounds__(256)
void gemm_wmma_f16_kernel(const float* __restrict__ A,
                          const float* __restrict__ Bm,
                          const float* __restrict__ bias,
                          const float* __restrict__ residual,
                          OutT* __restrict__ C,
                          int M, int N, int K)
{
    const int LDA = 36;
    __shared__ __align__(16) float    ldsA[2][128 * 36];
    __shared__ __align__(32) _Float16 ldsB[2][32 * 16];

    const int t     = threadIdx.x;
    const int lane  = t & 31;
    const int wave  = t >> 5;
    const int mtile = blockIdx.x * 128;
    const int ntile = blockIdx.y * 16;
    const int waveM = wave * 16;

    const float* Abase = A + (long)mtile * K;
    const int rows = M - mtile;                      // uniform; >=1
    const unsigned ldsAaddr[2] = {
        (unsigned)(unsigned long long)&ldsA[0][0],
        (unsigned)(unsigned long long)&ldsA[1][0]
    };

    // Prologue: kick off the first A tile DMA.
    if (wave == 0) tdm_load_a_tile(Abase, ldsAaddr[0], rows, K);

    v8f acc = {};

    #pragma unroll 2
    for (int kb = 0; kb < K; kb += 32) {
        const int cur = (kb >> 5) & 1;

        // ---- Stage B tile (32 x 16) as f16 into ldsB[cur]
        {
            int row = t >> 3;               // 0..31
            int col = (t & 7) << 1;         // 0,2,...,14
            const float2 f2 = *(const float2*)(Bm + (long)(kb + row) * N + ntile + col);
            ldsB[cur][row * 16 + col]     = (_Float16)f2.x;
            ldsB[cur][row * 16 + col + 1] = (_Float16)f2.y;
        }

        // ---- A tile (cur) DMA complete?
        if (wave == 0) __builtin_amdgcn_s_wait_tensorcnt(0);
        __syncthreads();

        // ---- Kick next A tile DMA into the other buffer (overlaps compute)
        if (wave == 0 && kb + 32 < K)
            tdm_load_a_tile(Abase + kb + 32, ldsAaddr[cur ^ 1], rows, K);

        // ---- A fragment (16x32 f16): lane holds row M = lane%16; K striped
        // over VGPR index and lane-half (ISA 7.12.2):
        //   K(vgpr v, half h, laneHalf) = (v/4)*16 + laneHalf*8 + (v%4)*2 + h
        const int mrow  = lane & 15;
        const int khalf = lane >> 4;
        v16h afrag = {};
        #pragma unroll
        for (int i = 0; i < 8; ++i) {
            int kk = ((i >> 2) << 4) + (khalf << 3) + ((i & 3) << 1);
            const float2 f2 = *(const float2*)&ldsA[cur][(waveM + mrow) * LDA + kk];
            afrag[2 * i]     = (_Float16)f2.x;
            afrag[2 * i + 1] = (_Float16)f2.y;
        }
        // B fragment (32x16 f16): lane holds column N = lane%16; K in same order as A
        v16h bfrag;
        #pragma unroll
        for (int s = 0; s < 16; ++s) {
            int kk = ((s >> 3) << 4) + (khalf << 3) + (s & 7);
            bfrag[s] = ldsB[cur][kk * 16 + mrow];
        }

        acc = __builtin_amdgcn_wmma_f32_16x16x32_f16(false, afrag, false, bfrag,
                                                     (short)0, acc, false, false);
        // No second barrier: buffers rotate, a wave can be at most one barrier
        // ahead and only ever touches the other buffer.
    }

    // D layout: VGPR r, lane l -> M = r + 8*(l/16), N = l%16
    const int n  = ntile + (lane & 15);
    const int mo = (lane >> 4) * 8;
    const float bn = bias ? bias[n] : 0.0f;
    const int mbase = mtile + waveM + mo;
    if (mtile + 128 <= M) {
        #pragma unroll
        for (int r = 0; r < 8; ++r) {
            float val = acc[r] + bn;
            if (residual) val += residual[(long)(mbase + r) * N + n];
            C[(long)(mbase + r) * N + n] = (OutT)val;
        }
    } else {
        #pragma unroll
        for (int r = 0; r < 8; ++r) {
            int m = mbase + r;
            if (m < M) {
                float val = acc[r] + bn;
                if (residual) val += residual[(long)m * N + n];
                C[(long)m * N + n] = (OutT)val;
            }
        }
    }
}

// ---------------------------------------------------------------------------
// MS deformable attention sampling: one wave per (b,q,h); lane = HD channel.
// v is f16 (80MB -> L2-resident on the 192MB L2). Branch-free corner gathers:
// clamped address always loaded, validity folded into the bilinear weight.
// ---------------------------------------------------------------------------
__global__ __launch_bounds__(256)
void msda_sample_kernel(const _Float16* __restrict__ v,    // [B,S,E] f16
                        const float* __restrict__ off,     // [B,Q,NH*NL*NP*2]
                        const float* __restrict__ logits,  // [B,Q,NH*NL*NP]
                        const float* __restrict__ ref,     // [B,Q,NL,2]
                        float* __restrict__ out_mid)       // [B,Q,E]
{
    const int lane = threadIdx.x & 31;
    const int w    = blockIdx.x * 8 + (threadIdx.x >> 5);
    const int h    = w % NH_;
    const int q    = (w / NH_) % Q_;
    const int b    = w / (NH_ * Q_);
    const int bq   = b * Q_ + q;

    // softmax over NL*NP = 16 logits (vectorized loads)
    const float4* lp4 = (const float4*)(logits + (long)bq * (NH_ * NL_ * NP_)
                                        + h * (NL_ * NP_));
    float li[16];
    #pragma unroll
    for (int i = 0; i < 4; ++i) {
        float4 f = lp4[i];
        li[4 * i + 0] = f.x; li[4 * i + 1] = f.y;
        li[4 * i + 2] = f.z; li[4 * i + 3] = f.w;
    }
    float mx = -1e30f;
    #pragma unroll
    for (int i = 0; i < 16; ++i) mx = fmaxf(mx, li[i]);
    float sum = 0.0f;
    #pragma unroll
    for (int i = 0; i < 16; ++i) { li[i] = __expf(li[i] - mx); sum += li[i]; }
    const float inv = 1.0f / sum;

    const int Hs[4] = {92, 46, 23, 12};
    const int Ws[4] = {160, 80, 40, 20};
    const int Ls[4] = {0, 14720, 18400, 19320};

    const _Float16* vb = v + (long)b * S_ * E_ + h * HD_ + lane;
    const float*    op = off + (long)bq * (NH_ * NL_ * NP_ * 2) + h * (NL_ * NP_ * 2);
    const float*    rp = ref + (long)bq * (NL_ * 2);

    float acc = 0.0f;
    #pragma unroll
    for (int l = 0; l < NL_; ++l) {
        const int Hl = Hs[l], Wl = Ws[l], st = Ls[l];
        const float2 r2 = *(const float2*)(rp + l * 2);
        #pragma unroll
        for (int p = 0; p < NP_; ++p) {
            const float2 o2 = *(const float2*)(op + (l * NP_ + p) * 2);
            // loc = ref + off/(W,H); pixel = loc*W - 0.5 (align_corners=False)
            const float x = (r2.x + o2.x / (float)Wl) * (float)Wl - 0.5f;
            const float y = (r2.y + o2.y / (float)Hl) * (float)Hl - 0.5f;
            const float x0f = floorf(x), y0f = floorf(y);
            const float wx = x - x0f, wy = y - y0f;
            const int   x0 = (int)x0f, y0 = (int)y0f;
            float s = 0.0f;
            #pragma unroll
            for (int cy = 0; cy < 2; ++cy) {
                #pragma unroll
                for (int cx = 0; cx < 2; ++cx) {
                    const int yi = y0 + cy;
                    const int xi = x0 + cx;
                    const float validf =
                        ((yi >= 0) & (yi < Hl) & (xi >= 0) & (xi < Wl)) ? 1.0f : 0.0f;
                    const int yc = yi < 0 ? 0 : (yi > Hl - 1 ? Hl - 1 : yi);
                    const int xc = xi < 0 ? 0 : (xi > Wl - 1 ? Wl - 1 : xi);
                    // always-in-bounds coalesced 64B gather (32 lanes x f16)
                    const float g = (float)vb[(long)(st + yc * Wl + xc) * E_];
                    const float wgt = (cy ? wy : 1.0f - wy) * (cx ? wx : 1.0f - wx);
                    s += g * (wgt * validf);
                }
            }
            acc += (li[l * NP_ + p] * inv) * s;
        }
    }
    out_mid[(long)bq * E_ + h * HD_ + lane] = acc;
}

// ---------------------------------------------------------------------------
extern "C" void kernel_launch(void* const* d_in, const int* in_sizes, int n_in,
                              void* d_out, int out_size, void* d_ws, size_t ws_size,
                              hipStream_t stream)
{
    (void)in_sizes; (void)n_in; (void)out_size; (void)ws_size;

    const float* query  = (const float*)d_in[0];
    const float* value  = (const float*)d_in[1];
    const float* ref    = (const float*)d_in[2];
    // d_in[3] = spatial_shapes (int32) -- baked in as compile-time constants
    const float* W_off  = (const float*)d_in[4];
    const float* b_off  = (const float*)d_in[5];
    const float* W_attn = (const float*)d_in[6];
    const float* b_attn = (const float*)d_in[7];
    const float* W_v    = (const float*)d_in[8];
    const float* b_v    = (const float*)d_in[9];
    const float* W_out  = (const float*)d_in[10];
    const float* b_out  = (const float*)d_in[11];
    float* out = (float*)d_out;

    char*      ws     = (char*)d_ws;
    _Float16*  v_ws   = (_Float16*)ws;                            // B*S*E (f16)
    float*     off_ws = (float*)(ws + (size_t)B_ * S_ * E_ * 2);  // B*Q*256
    float*     log_ws = off_ws + (size_t)B_ * Q_ * 256;           // B*Q*128
    float*     mid_ws = log_ws + (size_t)B_ * Q_ * 128;           // B*Q*256

    dim3 blk(256);

    // 1) v = value @ W_v + b_v  -> f16   [156480,256]x[256,256]
    {
        int M = B_ * S_;
        dim3 grd((M + 127) / 128, E_ / 16);
        gemm_wmma_f16_kernel<_Float16><<<grd, blk, 0, stream>>>(
            value, W_v, b_v, nullptr, v_ws, M, E_, E_);
    }
    // 2) off = query @ W_off + b_off     [7200,256]
    {
        int M = B_ * Q_;
        dim3 grd((M + 127) / 128, 256 / 16);
        gemm_wmma_f16_kernel<float><<<grd, blk, 0, stream>>>(
            query, W_off, b_off, nullptr, off_ws, M, 256, E_);
    }
    // 3) logits = query @ W_attn + b_attn  [7200,128]
    {
        int M = B_ * Q_;
        dim3 grd((M + 127) / 128, 128 / 16);
        gemm_wmma_f16_kernel<float><<<grd, blk, 0, stream>>>(
            query, W_attn, b_attn, nullptr, log_ws, M, 128, E_);
    }
    // 4) softmax + bilinear sampling (57600 waves, 8 per block)
    {
        dim3 grd((B_ * Q_ * NH_) / 8);
        msda_sample_kernel<<<grd, blk, 0, stream>>>(v_ws, off_ws, log_ws, ref, mid_ws);
    }
    // 5) out = mid @ W_out + b_out + query (residual)
    {
        int M = B_ * Q_;
        dim3 grd((M + 127) / 128, E_ / 16);
        gemm_wmma_f16_kernel<float><<<grd, blk, 0, stream>>>(
            mid_ws, W_out, b_out, query, out, M, E_, E_);
    }
}
